// ODEFunc_causal_att_44890998177846
// MI455X (gfx1250) — matrix operand order repaired and checked by reading, not compiled, not run on hardware
//
#include <hip/hip_runtime.h>
#include <math.h>

// ---------------------------------------------------------------------------
// ODEFunc fused pipeline for MI455X (gfx1250, wave32, WMMA)
//   B=64, D=H=2048
//   dz = tanh(z@W1+b1)@W2+b2
//   dK = exp(z@Wk+bk); v = z@Wv+bv; q = exp(z@Wq+bq)
//   dq = q * (dz@Wq+bq)
//   a  = sum(q*dK); da = a + sum(dq*K)
//   dV = v (outer) dK          (1 GiB stream write, NT)
//   dc = a*v + V @ dq          (1 GiB stream read, NT)   [dV@q == a*v exactly]
// ---------------------------------------------------------------------------

typedef __attribute__((ext_vector_type(16))) _Float16 v16h;
typedef __attribute__((ext_vector_type(8)))  float    v8f;
typedef __attribute__((ext_vector_type(4)))  float    f4;

#define BB 64
#define DD 2048
#define HH 2048

enum { EPI_NONE = 0, EPI_TANH = 1, EPI_EXP = 2, EPI_MUL = 3 };

// One 16x16 output tile per wave, K swept in steps of 32 via
// v_wmma_f32_16x16x32_f16 (f32 accumulate). fp32 A/W converted to f16 on load.
template <int OP>
__global__ __launch_bounds__(256)
void gemm_wmma_kernel(const float* __restrict__ A,      // [M,K] fp32
                      const float* __restrict__ W,      // [K,N] fp32 row-major
                      const float* __restrict__ bias,   // [N]
                      const float* __restrict__ mulsrc, // [M,N] (EPI_MUL) or null
                      float* __restrict__ out,          // [M,N]
                      int K, int N) {
  const int lane  = threadIdx.x & 31;
  const int wave  = threadIdx.x >> 5;
  const int ntile = blockIdx.x * 8 + wave;   // 8 waves per block, one N-tile each
  const int mtile = blockIdx.y;
  const int m     = lane & 15;
  const int hlf   = lane >> 4;               // lane half selects K sub-range

  const float* Arow = A + (size_t)(mtile * 16 + m) * K;

  v8f acc = {};
  for (int k0 = 0; k0 < K; k0 += 32) {
    v16h af, bf;
    // A fragment 16x32 f16: lanes 0-15 -> K {0..15}, lanes 16-31 -> K {16..31}
    const float* ap = Arow + k0 + hlf * 16;
    #pragma unroll
    for (int j = 0; j < 8; ++j) af[j] = (_Float16)ap[j];
    #pragma unroll
    for (int j = 0; j < 8; ++j) af[8 + j] = (_Float16)ap[8 + j];
    // B fragment 32x16 f16: lane holds column n = lane&15, contiguous K
    // range of 16 starting at hlf*16 (column of row-major W).
    const float* bp = W + (size_t)(k0 + hlf * 16) * N + ntile * 16 + (lane & 15);
    #pragma unroll
    for (int j = 0; j < 16; ++j) bf[j] = (_Float16)bp[(size_t)j * N];

    acc = __builtin_amdgcn_wmma_f32_16x16x32_f16(
        /*neg_a=*/false, af, /*neg_b=*/false, bf,
        /*c_mod=*/(short)0, acc, /*reuse_a=*/false, /*reuse_b=*/false);
  }

  // C/D layout: VGPR r -> row hlf*8 + r, col = lane&15
  const int   col = ntile * 16 + (lane & 15);
  const float bv  = bias[col];
  #pragma unroll
  for (int r = 0; r < 8; ++r) {
    const int row = mtile * 16 + hlf * 8 + r;
    float val = acc[r] + bv;
    if (OP == EPI_TANH) val = tanhf(val);
    if (OP == EPI_EXP)  val = __expf(val);
    if (OP == EPI_MUL)  val = mulsrc[(size_t)row * N + col] * val;
    out[(size_t)row * N + col] = val;
  }
}

// a[b] = sum_d q*dK ; da[b] = a[b] + sum_d dq*K
__global__ __launch_bounds__(256)
void reduce_a_da_kernel(const float* __restrict__ q,  const float* __restrict__ dK,
                        const float* __restrict__ dq, const float* __restrict__ Kin,
                        float* __restrict__ a_ws, float* __restrict__ da_out) {
  const int b = blockIdx.x;
  float s1 = 0.f, s2 = 0.f;
  for (int d = threadIdx.x; d < DD; d += 256) {
    const size_t i = (size_t)b * DD + d;
    s1 += q[i] * dK[i];
    s2 += dq[i] * Kin[i];
  }
  __shared__ float sm1[256], sm2[256];
  sm1[threadIdx.x] = s1; sm2[threadIdx.x] = s2;
  __syncthreads();
  for (int off = 128; off > 0; off >>= 1) {
    if (threadIdx.x < off) {
      sm1[threadIdx.x] += sm1[threadIdx.x + off];
      sm2[threadIdx.x] += sm2[threadIdx.x + off];
    }
    __syncthreads();
  }
  if (threadIdx.x == 0) {
    a_ws[b]   = sm1[0];
    da_out[b] = sm1[0] + sm2[0];
  }
}

// Fused streaming pass: one wave per (b, d) row.
//   dV[b,d,:] = v[b,d] * dK[b,:]          (NT store, 1 GiB)
//   dc[b,d]   = a[b]*v[b,d] + dot(V[b,d,:], dq[b,:])   (NT load of V, 1 GiB)
__global__ __launch_bounds__(256)
void outer_dc_kernel(const float* __restrict__ v,  const float* __restrict__ dK,
                     const float* __restrict__ dq, const float* __restrict__ Vin,
                     const float* __restrict__ a_ws,
                     float* __restrict__ dV, float* __restrict__ dc) {
  const int lane = threadIdx.x & 31;
  const int wave = threadIdx.x >> 5;
  const int b    = blockIdx.y;
  const int d    = blockIdx.x * 8 + wave;

  const float  vd     = v[(size_t)b * DD + d];
  const float  ab     = a_ws[b];
  const size_t rowoff = ((size_t)b * DD + d) * DD;
  const size_t broff  = (size_t)b * DD;

  float s = 0.f;
  for (int e = lane * 4; e < DD; e += 128) {
    f4 Vv  = __builtin_nontemporal_load((const f4*)(Vin + rowoff + e));
    f4 dKv = *(const f4*)(dK + broff + e);   // L1/L2 resident (8 KB per b)
    f4 dqv = *(const f4*)(dq + broff + e);
    f4 o   = vd * dKv;
    __builtin_nontemporal_store(o, (f4*)(dV + rowoff + e));
    s += Vv[0] * dqv[0] + Vv[1] * dqv[1] + Vv[2] * dqv[2] + Vv[3] * dqv[3];
  }
  // wave32 reduction
  #pragma unroll
  for (int off = 16; off > 0; off >>= 1) s += __shfl_down(s, off, 32);
  if (lane == 0) dc[(size_t)b * DD + d] = ab * vd + s;
}

extern "C" void kernel_launch(void* const* d_in, const int* in_sizes, int n_in,
                              void* d_out, int out_size, void* d_ws, size_t ws_size,
                              hipStream_t stream) {
  // inputs: 0:t 1:z 2:K 3:V 4:Wq 5:bq 6:Wk 7:bk 8:Wv 9:bv 10:W1 11:b1 12:W2 13:b2
  const float* z   = (const float*)d_in[1];
  const float* Kin = (const float*)d_in[2];
  const float* Vin = (const float*)d_in[3];
  const float* Wq  = (const float*)d_in[4];
  const float* bq  = (const float*)d_in[5];
  const float* Wk  = (const float*)d_in[6];
  const float* bk  = (const float*)d_in[7];
  const float* Wv  = (const float*)d_in[8];
  const float* bv  = (const float*)d_in[9];
  const float* W1  = (const float*)d_in[10];
  const float* b1  = (const float*)d_in[11];
  const float* W2  = (const float*)d_in[12];
  const float* b2  = (const float*)d_in[13];

  // output layout (fp32, concatenated): dz, da, dc, dK, dV
  float* out = (float*)d_out;
  float* dz  = out;                       // 131072
  float* da  = out + 131072;              // 64
  float* dc  = out + 131136;              // 131072
  float* dK  = out + 262208;              // 131072
  float* dV  = out + 393280;              // 268435456

  // workspace (fp32): h, q, v, dq, a  (~2.1 MB)
  float* ws = (float*)d_ws;
  float* h  = ws;
  float* q  = ws + 131072;
  float* v  = ws + 262144;
  float* dq = ws + 393216;
  float* a  = ws + 524288;

  dim3 gblk(256);
  dim3 ggrid(DD / 16 / 8, BB / 16);       // (16, 4): 128 N-tiles x 4 M-tiles

  gemm_wmma_kernel<EPI_TANH><<<ggrid, gblk, 0, stream>>>(z,  W1, b1, nullptr, h,  DD, HH);
  gemm_wmma_kernel<EPI_NONE><<<ggrid, gblk, 0, stream>>>(h,  W2, b2, nullptr, dz, HH, DD);
  gemm_wmma_kernel<EPI_EXP ><<<ggrid, gblk, 0, stream>>>(z,  Wk, bk, nullptr, dK, DD, DD);
  gemm_wmma_kernel<EPI_NONE><<<ggrid, gblk, 0, stream>>>(z,  Wv, bv, nullptr, v,  DD, DD);
  gemm_wmma_kernel<EPI_EXP ><<<ggrid, gblk, 0, stream>>>(z,  Wq, bq, nullptr, q,  DD, DD);
  gemm_wmma_kernel<EPI_MUL ><<<ggrid, gblk, 0, stream>>>(dz, Wq, bq, q,       dq, DD, DD);

  reduce_a_da_kernel<<<BB, 256, 0, stream>>>(q, dK, dq, Kin, a, da);

  dim3 ogrid(DD / 8, BB);                 // (256, 64): 8 rows per block
  outer_dc_kernel<<<ogrid, 256, 0, stream>>>(v, dK, dq, Vin, a, dV, dc);
}
